// EncoderLayer_23003844837391
// MI455X (gfx1250) — matrix operand order repaired and checked by reading, not compiled, not run on hardware
//
#include <hip/hip_runtime.h>
#include <hip/hip_bf16.h>
#include <math.h>

// ---------------------------------------------------------------------------
// EncoderLayer for MI455X (gfx1250): bf16 WMMA GEMMs + f32 softmax/LN/top-k
// B=4, L=960 (512+256+128+64), D=512, H=8, Dk=Dv=64, D_inner=2048, TEMP=8
// ---------------------------------------------------------------------------

typedef __attribute__((ext_vector_type(16))) __bf16 v16bf;
typedef __attribute__((ext_vector_type(8)))  float  v8f;

#define BB   4
#define LL   960
#define DD   512
#define HH   8
#define DK   64
#define DI   2048
#define NROW (BB*LL)      // 3840
#define TOPK 10
#define LDS_STRIDE 40     // elements; 80B rows -> every 8-elem chunk 16B aligned

// ---- WMMA fragment loaders (layouts per CDNA5 ISA §7.12.2) -----
// A (16x32 bf16), LDS layout As[m][k] stride 40:
// lane m=lane&15, hi=lane>>4: elems = As[m][8*hi .. 8*hi+7] ++ As[m][16+8*hi .. +7]
__device__ __forceinline__ v16bf frag_a(const __bf16* As, int lane, int moff) {
  int m  = (lane & 15) + moff;
  int hi = lane >> 4;
  const __bf16* row = As + m * LDS_STRIDE;
  v16bf a;
#pragma unroll
  for (int e = 0; e < 8; ++e)  a[e]     = row[8 * hi + e];
#pragma unroll
  for (int e = 0; e < 8; ++e)  a[8 + e] = row[16 + 8 * hi + e];
  return a;
}
// B (32x16 bf16), LDS layout TRANSPOSED Bs[n][k] stride 40:
// lane n=lane&15, kh=lane>>4: elems = Bs[n][8*kh .. 8*kh+7] ++ Bs[n][16+8*kh .. +7]
__device__ __forceinline__ v16bf frag_b(const __bf16* Bs, int lane, int noff) {
  const __bf16* row = Bs + ((lane & 15) + noff) * LDS_STRIDE;
  int kh = lane >> 4;
  v16bf b;
#pragma unroll
  for (int e = 0; e < 8; ++e) { b[e] = row[8 * kh + e]; b[8 + e] = row[16 + 8 * kh + e]; }
  return b;
}

__device__ __forceinline__ void store8_bf16(__bf16* dst, const __bf16* t) {
  *(uint4*)dst = *(const uint4*)t;
}

// ---------------------------------------------------------------------------
// Generic GEMM: C[M,N] = A[M,K] @ B[K,N] (+bias)(+residual)(relu?) -> f32/bf16
// Block tile 64x64, 8 waves (2 x 4), each wave two 16x16 WMMA accumulators.
// Templated on A element type (float converts on stage; bf16 copies 16B).
// ---------------------------------------------------------------------------
template <bool ABF16>
__global__ __launch_bounds__(256) void gemm_wmma(
    const void* __restrict__ Aptr, int lda,
    const float* __restrict__ Bw, int ldb,
    const float* __restrict__ bias,
    const float* __restrict__ resid,
    int relu,
    float* __restrict__ outF, __bf16* __restrict__ outB,
    int N, int K)
{
  __shared__ __bf16 As[64 * LDS_STRIDE];
  __shared__ __bf16 Bs[64 * LDS_STRIDE];
  int tid = threadIdx.x, lane = tid & 31, wid = tid >> 5;
  int waveM = wid >> 2, waveN = wid & 3;
  int m0 = blockIdx.y * 64, n0 = blockIdx.x * 64;
  const float*  Af = (const float*)Aptr;
  const __bf16* Ab = (const __bf16*)Aptr;
  v8f c0 = {}; v8f c1 = {};

  int am = tid >> 2;            // 0..63  A row
  int ak = (tid & 3) * 8;       // 0,8,16,24
  int bn = tid & 63;            // 0..63  B col (LDS row)
  int bk = (tid >> 6) * 8;      // 0,8,16,24

  for (int kc = 0; kc < K; kc += 32) {
    // stage A tile (64x32): one 16B chunk per thread
    if (ABF16) {
      const __bf16* src = Ab + (size_t)(m0 + am) * lda + kc + ak;
      store8_bf16(&As[am * LDS_STRIDE + ak], src);
    } else {
      const float* src = Af + (size_t)(m0 + am) * lda + kc + ak;
      float4 f0 = *(const float4*)src;
      float4 f1 = *(const float4*)(src + 4);
      alignas(16) __bf16 t[8] = {(__bf16)f0.x, (__bf16)f0.y, (__bf16)f0.z, (__bf16)f0.w,
                                 (__bf16)f1.x, (__bf16)f1.y, (__bf16)f1.z, (__bf16)f1.w};
      store8_bf16(&As[am * LDS_STRIDE + ak], t);
    }
    // stage B tile (32x64) transposed -> Bs[n][k]; coalesced dword gathers
    {
      alignas(16) __bf16 t[8];
#pragma unroll
      for (int u = 0; u < 8; ++u)
        t[u] = (__bf16)Bw[(size_t)(kc + bk + u) * ldb + n0 + bn];
      store8_bf16(&Bs[bn * LDS_STRIDE + bk], t);
    }
    __syncthreads();
    v16bf a0 = frag_a(As, lane, waveM * 32);
    v16bf a1 = frag_a(As, lane, waveM * 32 + 16);
    v16bf b  = frag_b(Bs, lane, waveN * 16);
    c0 = __builtin_amdgcn_wmma_f32_16x16x32_bf16(false, a0, false, b, (short)0, c0, false, false);
    c1 = __builtin_amdgcn_wmma_f32_16x16x32_bf16(false, a1, false, b, (short)0, c1, false, false);
    __syncthreads();
  }

  int hi = lane >> 4;
  int col = n0 + waveN * 16 + (lane & 15);
  float bv = bias ? bias[col] : 0.0f;
#pragma unroll
  for (int r = 0; r < 8; ++r) {
    int row0 = m0 + waveM * 32 + r + 8 * hi;
    int row1 = row0 + 16;
    float v0 = c0[r] + bv;
    float v1 = c1[r] + bv;
    if (resid) { v0 += resid[(size_t)row0 * N + col]; v1 += resid[(size_t)row1 * N + col]; }
    if (relu)  { v0 = fmaxf(v0, 0.f); v1 = fmaxf(v1, 0.f); }
    if (outF)  { outF[(size_t)row0 * N + col] = v0; outF[(size_t)row1 * N + col] = v1; }
    if (outB)  { outB[(size_t)row0 * N + col] = (__bf16)v0; outB[(size_t)row1 * N + col] = (__bf16)v1; }
  }
}

// ---------------------------------------------------------------------------
// Attention scores: S[b,h,q,k] = scale * sum_d Q[b,q,h,d] K[b,k,h,d]  (WMMA)
// Both Q and K tiles are d-contiguous in memory -> pure 16B copy staging.
// ---------------------------------------------------------------------------
__global__ __launch_bounds__(256) void scores_wmma(
    const __bf16* __restrict__ q, const __bf16* __restrict__ kbuf,
    float* __restrict__ attn, float scale)
{
  __shared__ __bf16 As[64 * LDS_STRIDE];
  __shared__ __bf16 Bs[64 * LDS_STRIDE];
  int tid = threadIdx.x, lane = tid & 31, wid = tid >> 5;
  int waveM = wid >> 2, waveN = wid & 3;
  int b = blockIdx.z >> 3, h = blockIdx.z & 7;
  int m0 = blockIdx.y * 64, n0 = blockIdx.x * 64;
  size_t base = ((size_t)b * LL) * DD + (size_t)h * DK;   // [b, l, h, d]
  v8f c0 = {}; v8f c1 = {};

  int am = tid >> 2;
  int ak = (tid & 3) * 8;

  for (int kc = 0; kc < DK; kc += 32) {
    store8_bf16(&As[am * LDS_STRIDE + ak], q    + base + (size_t)(m0 + am) * DD + kc + ak);
    store8_bf16(&Bs[am * LDS_STRIDE + ak], kbuf + base + (size_t)(n0 + am) * DD + kc + ak);
    __syncthreads();
    v16bf a0 = frag_a(As, lane, waveM * 32);
    v16bf a1 = frag_a(As, lane, waveM * 32 + 16);
    v16bf bb = frag_b(Bs, lane, waveN * 16);
    c0 = __builtin_amdgcn_wmma_f32_16x16x32_bf16(false, a0, false, bb, (short)0, c0, false, false);
    c1 = __builtin_amdgcn_wmma_f32_16x16x32_bf16(false, a1, false, bb, (short)0, c1, false, false);
    __syncthreads();
  }

  int hi = lane >> 4;
  int col = n0 + waveN * 16 + (lane & 15);
  size_t obase = (((size_t)b * HH + h) * LL) * (size_t)LL;
#pragma unroll
  for (int r = 0; r < 8; ++r) {
    int row0 = m0 + waveM * 32 + r + 8 * hi;
    attn[obase + (size_t)row0 * LL + col]        = c0[r] * scale;
    attn[obase + (size_t)(row0 + 16) * LL + col] = c1[r] * scale;
  }
}

// ---------------------------------------------------------------------------
// In-place masked softmax over rows of attn; mask = top-10 similar columns.
// ---------------------------------------------------------------------------
__global__ __launch_bounds__(128) void softmax_mask(
    float* __restrict__ attn, const int* __restrict__ sem_idx, int use_mask)
{
  int row = blockIdx.x;                 // (b*H + h)*L + q
  int q = row % LL;
  int b = row / (HH * LL);
  float* p = attn + (size_t)row * LL;
  int tid = threadIdx.x;
  int midx[TOPK];
  if (use_mask) {
    const int* mi = sem_idx + ((size_t)b * LL + q) * TOPK;
#pragma unroll
    for (int t = 0; t < TOPK; ++t) midx[t] = mi[t];
  } else {
#pragma unroll
    for (int t = 0; t < TOPK; ++t) midx[t] = -1;
  }
  __shared__ float red[128];

  float mx = -3.4e38f;
  for (int c = tid; c < LL; c += 128) {
    bool msk = false;
#pragma unroll
    for (int t = 0; t < TOPK; ++t) msk = msk || (c == midx[t]);
    if (!msk) mx = fmaxf(mx, p[c]);
  }
  red[tid] = mx; __syncthreads();
  for (int s = 64; s > 0; s >>= 1) {
    if (tid < s) red[tid] = fmaxf(red[tid], red[tid + s]);
    __syncthreads();
  }
  mx = red[0]; __syncthreads();

  float sum = 0.f;
  for (int c = tid; c < LL; c += 128) {
    bool msk = false;
#pragma unroll
    for (int t = 0; t < TOPK; ++t) msk = msk || (c == midx[t]);
    if (!msk) sum += __expf(p[c] - mx);
  }
  red[tid] = sum; __syncthreads();
  for (int s = 64; s > 0; s >>= 1) {
    if (tid < s) red[tid] += red[tid + s];
    __syncthreads();
  }
  float inv = 1.0f / red[0]; __syncthreads();

  for (int c = tid; c < LL; c += 128) {
    bool msk = false;
#pragma unroll
    for (int t = 0; t < TOPK; ++t) msk = msk || (c == midx[t]);
    p[c] = msk ? 0.f : __expf(p[c] - mx) * inv;
  }
}

// ---------------------------------------------------------------------------
// P @ V: heads[b,q,h,d] = sum_k P[b,h,q,k] V[b,k,h,d]  (WMMA, K=960, N=64)
// ---------------------------------------------------------------------------
__global__ __launch_bounds__(256) void pv_wmma(
    const float* __restrict__ probs, const __bf16* __restrict__ vbuf,
    float* __restrict__ heads)
{
  __shared__ __bf16 As[64 * LDS_STRIDE];
  __shared__ __bf16 Bs[64 * LDS_STRIDE];
  int tid = threadIdx.x, lane = tid & 31, wid = tid >> 5;
  int waveM = wid >> 2, waveN = wid & 3;
  int b = blockIdx.z >> 3, h = blockIdx.z & 7;
  int m0 = blockIdx.y * 64;
  size_t pbase = (((size_t)b * HH + h) * LL) * (size_t)LL;
  size_t vbase = ((size_t)b * LL) * DD + (size_t)h * DK;
  v8f c0 = {}; v8f c1 = {};

  int am = tid >> 2;
  int ak = (tid & 3) * 8;
  int bn = tid & 63;            // d within head
  int bk = (tid >> 6) * 8;      // key sub-chunk

  for (int kc = 0; kc < LL; kc += 32) {
    {
      const float* src = probs + pbase + (size_t)(m0 + am) * LL + kc + ak;
      float4 f0 = *(const float4*)src;
      float4 f1 = *(const float4*)(src + 4);
      alignas(16) __bf16 t[8] = {(__bf16)f0.x, (__bf16)f0.y, (__bf16)f0.z, (__bf16)f0.w,
                                 (__bf16)f1.x, (__bf16)f1.y, (__bf16)f1.z, (__bf16)f1.w};
      store8_bf16(&As[am * LDS_STRIDE + ak], t);
    }
    {
      alignas(16) __bf16 t[8];
#pragma unroll
      for (int u = 0; u < 8; ++u)
        t[u] = vbuf[vbase + (size_t)(kc + bk + u) * DD + bn];
      store8_bf16(&Bs[bn * LDS_STRIDE + bk], t);
    }
    __syncthreads();
    v16bf a0 = frag_a(As, lane, waveM * 32);
    v16bf a1 = frag_a(As, lane, waveM * 32 + 16);
    v16bf bb = frag_b(Bs, lane, waveN * 16);
    c0 = __builtin_amdgcn_wmma_f32_16x16x32_bf16(false, a0, false, bb, (short)0, c0, false, false);
    c1 = __builtin_amdgcn_wmma_f32_16x16x32_bf16(false, a1, false, bb, (short)0, c1, false, false);
    __syncthreads();
  }

  int hi = lane >> 4;
  int col = waveN * 16 + (lane & 15);              // 0..63 = d
#pragma unroll
  for (int r = 0; r < 8; ++r) {
    int row0 = m0 + waveM * 32 + r + 8 * hi;
    heads[((size_t)b * LL + row0) * DD + h * DK + col]      = c0[r];
    heads[((size_t)b * LL + row0 + 16) * DD + h * DK + col] = c1[r];
  }
}

// ---------------------------------------------------------------------------
// Row LayerNorm over D=512 (one block per row, 2 elems/thread)
// ---------------------------------------------------------------------------
__global__ __launch_bounds__(256) void layernorm_k(
    const float* __restrict__ x, const float* __restrict__ g,
    const float* __restrict__ bta, float* __restrict__ y)
{
  int row = blockIdx.x, tid = threadIdx.x;
  const float* xr = x + (size_t)row * DD;
  float a0 = xr[tid], a1 = xr[tid + 256];
  __shared__ float red[256];
  red[tid] = a0 + a1; __syncthreads();
  for (int s = 128; s > 0; s >>= 1) {
    if (tid < s) red[tid] += red[tid + s];
    __syncthreads();
  }
  float mean = red[0] * (1.0f / DD); __syncthreads();
  float d0 = a0 - mean, d1 = a1 - mean;
  red[tid] = d0 * d0 + d1 * d1; __syncthreads();
  for (int s = 128; s > 0; s >>= 1) {
    if (tid < s) red[tid] += red[tid + s];
    __syncthreads();
  }
  float inv = rsqrtf(red[0] * (1.0f / DD) + 1e-6f);
  y[(size_t)row * DD + tid]       = d0 * inv * g[tid] + bta[tid];
  y[(size_t)row * DD + tid + 256] = d1 * inv * g[tid + 256] + bta[tid + 256];
}

// enc = (a + b + c) / 3
__global__ void combine_k(const float* __restrict__ a, const float* __restrict__ b,
                          const float* __restrict__ c, float* __restrict__ o, int n)
{
  int i = blockIdx.x * blockDim.x + threadIdx.x;
  if (i < n) o[i] = (a[i] + b[i] + c[i]) * (1.0f / 3.0f);
}

// ---------------------------------------------------------------------------
// Semantic top-k: for each (b, row i) find 10 nearest rows in its block
// (sim = -euclid dist), store absolute column indices (these get -inf mask).
// ---------------------------------------------------------------------------
__global__ __launch_bounds__(256) void sem_topk(
    const float* __restrict__ x, int* __restrict__ sem_idx)
{
  int i = blockIdx.x, b = blockIdx.y, tid = threadIdx.x;
  int off, n;
  if (i < 512)      { off = 0;   n = 512; }
  else if (i < 768) { off = 512; n = 256; }
  else if (i < 896) { off = 768; n = 128; }
  else              { off = 896; n = 64;  }

  __shared__ float xi[DD];
  __shared__ float sims[512];
  __shared__ float rv[256];
  __shared__ int   ri[256];

  const float* xrow = x + ((size_t)b * LL + i) * DD;
  xi[tid] = xrow[tid]; xi[tid + 256] = xrow[tid + 256];
  __syncthreads();
  float s = xi[tid] * xi[tid] + xi[tid + 256] * xi[tid + 256];
  rv[tid] = s; __syncthreads();
  for (int st = 128; st > 0; st >>= 1) {
    if (tid < st) rv[tid] += rv[tid + st];
    __syncthreads();
  }
  float sqi = rv[0]; __syncthreads();

  for (int j = tid; j < n; j += 256) {
    const float* xj = x + ((size_t)b * LL + off + j) * DD;
    float dot = 0.f, sqj = 0.f;
    for (int d = 0; d < DD; ++d) { float v = xj[d]; dot += xi[d] * v; sqj += v * v; }
    float d2 = sqi + sqj - 2.f * dot;
    sims[j] = -sqrtf(fmaxf(d2, 0.f));
  }
  __syncthreads();

  int* out = sem_idx + ((size_t)b * LL + i) * TOPK;
  for (int r = 0; r < TOPK; ++r) {
    float bv = -3.4e38f; int bi = 0x7fffffff;
    for (int j = tid; j < n; j += 256) {
      float v = sims[j];
      if (v > bv || (v == bv && j < bi)) { bv = v; bi = j; }
    }
    rv[tid] = bv; ri[tid] = bi; __syncthreads();
    for (int st = 128; st > 0; st >>= 1) {
      if (tid < st) {
        if (rv[tid + st] > rv[tid] ||
            (rv[tid + st] == rv[tid] && ri[tid + st] < ri[tid])) {
          rv[tid] = rv[tid + st]; ri[tid] = ri[tid + st];
        }
      }
      __syncthreads();
    }
    if (tid == 0) { out[r] = off + ri[0]; sims[ri[0]] = -3.4e38f; }
    __syncthreads();
  }
}

// ---------------------------------------------------------------------------
extern "C" void kernel_launch(void* const* d_in, const int* in_sizes, int n_in,
                              void* d_out, int out_size, void* d_ws, size_t ws_size,
                              hipStream_t stream) {
  (void)in_sizes; (void)n_in; (void)out_size; (void)ws_size;
  const float* x      = (const float*)d_in[0];
  // d_in[1] = all_size (constants baked in)
  const float* t_wq   = (const float*)d_in[2];
  const float* t_wk   = (const float*)d_in[3];
  const float* t_wv   = (const float*)d_in[4];
  const float* t_fc_w = (const float*)d_in[5];
  const float* t_fc_b = (const float*)d_in[6];
  const float* t_ln_g = (const float*)d_in[7];
  const float* t_ln_b = (const float*)d_in[8];
  const float* s_wq   = (const float*)d_in[9];
  const float* s_wk   = (const float*)d_in[10];
  const float* s_wv   = (const float*)d_in[11];
  const float* s_fc_w = (const float*)d_in[12];
  const float* s_fc_b = (const float*)d_in[13];
  const float* s_ln_g = (const float*)d_in[14];
  const float* s_ln_b = (const float*)d_in[15];
  const float* f_w1   = (const float*)d_in[16];
  const float* f_b1   = (const float*)d_in[17];
  const float* f_w2   = (const float*)d_in[18];
  const float* f_b2   = (const float*)d_in[19];
  const float* f_ln_g = (const float*)d_in[20];
  const float* f_ln_b = (const float*)d_in[21];

  float* out_enc = (float*)d_out;
  float* attn    = out_enc + (size_t)NROW * DD;      // [B,H,L,L] region of d_out

  char* ws = (char*)d_ws;
  size_t off = 0;
  auto alloc = [&](size_t bytes) -> void* {
    void* p = ws + off;
    off += (bytes + 255) & ~(size_t)255;
    return p;
  };
  float*  qn    = (float*) alloc((size_t)NROW * DD * 4);
  __bf16* qb    = (__bf16*)alloc((size_t)NROW * DD * 2);
  __bf16* kb    = (__bf16*)alloc((size_t)NROW * DD * 2);
  __bf16* vb    = (__bf16*)alloc((size_t)NROW * DD * 2);
  float*  heads = (float*) alloc((size_t)NROW * DD * 4);
  float*  outt  = (float*) alloc((size_t)NROW * DD * 4);
  float*  outs  = (float*) alloc((size_t)NROW * DD * 4);
  float*  encb  = (float*) alloc((size_t)NROW * DD * 4);
  __bf16* hb    = (__bf16*)alloc((size_t)NROW * DI * 2);
  int*    sidx  = (int*)   alloc((size_t)NROW * TOPK * 4);

  dim3 gProj(DD / 64, NROW / 64);          // (8, 60)
  dim3 gAttn(LL / 64, LL / 64, BB * HH);   // (15, 15, 32)
  dim3 gPV(1, LL / 64, BB * HH);           // (1, 15, 32)

  // semantic mask indices (independent of MHAs)
  sem_topk<<<dim3(LL, BB), 256, 0, stream>>>(x, sidx);

  // ---------------- temporal MHA ----------------
  layernorm_k<<<NROW, 256, 0, stream>>>(x, t_ln_g, t_ln_b, qn);
  gemm_wmma<false><<<gProj, 256, 0, stream>>>(qn, DD, t_wq, DD, nullptr, nullptr, 0, nullptr, qb, DD, DD);
  gemm_wmma<false><<<gProj, 256, 0, stream>>>(x,  DD, t_wk, DD, nullptr, nullptr, 0, nullptr, kb, DD, DD);
  gemm_wmma<false><<<gProj, 256, 0, stream>>>(x,  DD, t_wv, DD, nullptr, nullptr, 0, nullptr, vb, DD, DD);
  scores_wmma<<<gAttn, 256, 0, stream>>>(qb, kb, attn, 0.125f);     // attn region as scratch
  softmax_mask<<<BB * HH * LL, 128, 0, stream>>>(attn, sidx, 0);
  pv_wmma<<<gPV, 256, 0, stream>>>(attn, vb, heads);
  gemm_wmma<false><<<gProj, 256, 0, stream>>>(heads, DD, t_fc_w, DD, t_fc_b, x, 0, outt, nullptr, DD, DD);

  // ---------------- semantic MHA ----------------
  layernorm_k<<<NROW, 256, 0, stream>>>(x, s_ln_g, s_ln_b, qn);
  gemm_wmma<false><<<gProj, 256, 0, stream>>>(qn, DD, s_wq, DD, nullptr, nullptr, 0, nullptr, qb, DD, DD);
  gemm_wmma<false><<<gProj, 256, 0, stream>>>(x,  DD, s_wk, DD, nullptr, nullptr, 0, nullptr, kb, DD, DD);
  gemm_wmma<false><<<gProj, 256, 0, stream>>>(x,  DD, s_wv, DD, nullptr, nullptr, 0, nullptr, vb, DD, DD);
  scores_wmma<<<gAttn, 256, 0, stream>>>(qb, kb, attn, 0.125f);
  softmax_mask<<<BB * HH * LL, 128, 0, stream>>>(attn, sidx, 1);    // final attn output
  pv_wmma<<<gPV, 256, 0, stream>>>(attn, vb, heads);
  gemm_wmma<false><<<gProj, 256, 0, stream>>>(heads, DD, s_fc_w, DD, s_fc_b, x, 0, outs, nullptr, DD, DD);

  // ---------------- combine + FFN ----------------
  combine_k<<<(NROW * DD) / 256, 256, 0, stream>>>(outt, outs, x, encb, NROW * DD);
  layernorm_k<<<NROW, 256, 0, stream>>>(encb, f_ln_g, f_ln_b, qn);
  gemm_wmma<false><<<dim3(DI / 64, NROW / 64), 256, 0, stream>>>(qn, DD, f_w1, DI, f_b1, nullptr, 1, nullptr, hb, DI, DD);
  gemm_wmma<true ><<<dim3(DD / 64, NROW / 64), 256, 0, stream>>>(hb, DI, f_w2, DD, f_b2, encb, 0, out_enc, nullptr, DD, DI);
}
